// RIConv2SetAbstraction_11330123727107
// MI455X (gfx1250) — matrix operand order repaired and checked by reading, not compiled, not run on hardware
//
#include <hip/hip_runtime.h>

typedef __attribute__((ext_vector_type(16))) _Float16 v16h;
typedef __attribute__((ext_vector_type(8)))  _Float16 v8h;
typedef __attribute__((ext_vector_type(8)))  float    v8f;

#define BATCH 16
#define SDIM  512
#define KDIM  32
#define NPTS  2048
#define NGRP  (BATCH * SDIM)   // 8192

// ---------------------------------------------------------------------------
// Kernel 0: fold BN into weights (f32 -> f16) and biases (f32), once per launch
//   y = g*(Wx+b)/sqrt(1+eps) + bt  =>  W' = W*g*sc ; bias' = b*g*sc + bt
// W1 is padded from [64][8] to [64][32] with zeros so layer-1 is a single K=32 WMMA.
// ---------------------------------------------------------------------------
__global__ void prep_kernel(
    const float* __restrict__ w1, const float* __restrict__ b1, const float* __restrict__ g1, const float* __restrict__ bt1,
    const float* __restrict__ w2, const float* __restrict__ b2, const float* __restrict__ g2, const float* __restrict__ bt2,
    const float* __restrict__ w3, const float* __restrict__ b3, const float* __restrict__ g3, const float* __restrict__ bt3,
    const float* __restrict__ w4, const float* __restrict__ b4, const float* __restrict__ g4, const float* __restrict__ bt4,
    _Float16* __restrict__ W1, _Float16* __restrict__ W2,
    _Float16* __restrict__ W3, _Float16* __restrict__ W4,
    float* __restrict__ BIAS)
{
    const float sc = rsqrtf(1.0f + 1e-5f);
    const int tid  = blockIdx.x * blockDim.x + threadIdx.x;
    const int nthr = gridDim.x * blockDim.x;

    for (int i = tid; i < 64 * 32; i += nthr) {
        int o = i >> 5, k = i & 31;
        W1[i] = (_Float16)((k < 8) ? w1[o * 8 + k] * g1[o] * sc : 0.0f);
    }
    for (int i = tid; i < 128 * 64;  i += nthr) { int o = i >> 6; W2[i] = (_Float16)(w2[i] * g2[o] * sc); }
    for (int i = tid; i < 128 * 128; i += nthr) { int o = i >> 7; W3[i] = (_Float16)(w3[i] * g3[o] * sc); }
    for (int i = tid; i < 256 * 128; i += nthr) { int o = i >> 7; W4[i] = (_Float16)(w4[i] * g4[o] * sc); }

    for (int i = tid; i < 64;  i += nthr) BIAS[i]       = b1[i] * g1[i] * sc + bt1[i];
    for (int i = tid; i < 128; i += nthr) BIAS[64 + i]  = b2[i] * g2[i] * sc + bt2[i];
    for (int i = tid; i < 128; i += nthr) BIAS[192 + i] = b3[i] * g3[i] * sc + bt3[i];
    for (int i = tid; i < 256; i += nthr) BIAS[320 + i] = b4[i] * g4[i] * sc + bt4[i];
}

// ---------------------------------------------------------------------------
// Kernel 1: geometry. One wave32 per (b,s) group; lane == neighbor k.
// Produces ri features as f16 in layout X0[group][k(sorted)][8].
// ---------------------------------------------------------------------------
__device__ __forceinline__ float clip_ac(float x) {
    return fminf(fmaxf(x, -1.0f + 1e-7f), 1.0f - 1e-7f);
}

__global__ __launch_bounds__(32) void geom_kernel(
    const float* __restrict__ xyz,          // [B][S][K][3]
    const float* __restrict__ centers,      // [B][S][3]
    const float* __restrict__ center_norms, // [B][S][3]
    const int*   __restrict__ idx,          // [B][S][K]
    const float* __restrict__ pts,          // [B][N][3]
    const float* __restrict__ norm_ori,     // [B][N][3]
    _Float16*    __restrict__ X0)           // [G][32][8] f16
{
    const int g    = blockIdx.x;        // b*S + s
    const int lane = threadIdx.x;       // k
    const int b    = g >> 9;            // S = 512

    const float* xp = xyz + ((size_t)g * KDIM + lane) * 3;
    float x0 = xp[0], x1 = xp[1], x2 = xp[2];
    const float* np_ = center_norms + (size_t)g * 3;
    float n0 = np_[0], n1 = np_[1], n2 = np_[2];
    const float* cp = centers + (size_t)g * 3;
    float c0 = cp[0], c1 = cp[1], c2 = cp[2];

    // project onto tangent plane of center normal
    float dist = x0 * n0 + x1 * n1 + x2 * n2;
    float p0 = x0 - dist * n0, p1 = x1 - dist * n1, p2 = x2 - dist * n2;
    float plen = sqrtf(p0 * p0 + p1 * p1 + p2 * p2);
    float pinv = (plen > 0.0f) ? (1.0f / plen) : 0.0f;
    float u0 = p0 * pinv, u1 = p1 * pinv, u2 = p2 * pinv;

    // wave argmax of proj_len (first occurrence on ties)
    float mv = plen; int mi = lane;
    #pragma unroll
    for (int off = 16; off > 0; off >>= 1) {
        float ov = __shfl_xor(mv, off, 32);
        int   oi = __shfl_xor(mi, off, 32);
        if (ov > mv || (ov == mv && oi < mi)) { mv = ov; mi = oi; }
    }
    float r0 = __shfl(u0, mi, 32), r1 = __shfl(u1, mi, 32), r2 = __shfl(u2, mi, 32);

    float dotv = u0 * r0 + u1 * r1 + u2 * r2;
    float cx = u1 * r2 - u2 * r1;
    float cy = u2 * r0 - u0 * r2;
    float cz = u0 * r1 - u1 * r0;
    float sd = cx * n0 + cy * n1 + cz * n2;
    float sgn = (sd > 0.0f) ? 1.0f : ((sd < 0.0f) ? -1.0f : 0.0f);
    if (lane == 0) sgn = 1.0f;
    float dots = sgn * dotv - (1.0f - sgn);

    // stable descending rank of dots across the wave
    int rank = 0;
    for (int j = 0; j < 32; ++j) {
        float dj = __shfl(dots, j, 32);
        if (dj > dots || (dj == dots && j < lane)) rank++;
    }

    __shared__ float sdots[32];
    __shared__ int   sidx[32];
    sdots[rank] = dots;
    sidx[rank]  = idx[(size_t)g * KDIM + lane];
    __syncthreads();

    float dsort = sdots[lane];
    int   io    = sidx[lane];
    float pia   = (lane == 0) ? (-3.0f - sdots[31]) : (dsort - sdots[lane - 1]);

    // gather in sorted order
    const float* P = pts + ((size_t)b * NPTS + io) * 3;
    float l0 = P[0] - c0, l1 = P[1] - c1, l2 = P[2] - c2;
    float llen = sqrtf(l0 * l0 + l1 * l1 + l2 * l2);
    float linv = (llen > 0.0f) ? (1.0f / llen) : 0.0f;
    float lu0 = l0 * linv, lu1 = l1 * linv, lu2 = l2 * linv;

    const float* GN = norm_ori + ((size_t)b * NPTS + io) * 3;
    float gn0 = GN[0], gn1 = GN[1], gn2 = GN[2];

    float a0 = lu0 * n0 + lu1 * n1 + lu2 * n2;
    float a1 = lu0 * gn0 + lu1 * gn1 + lu2 * gn2;
    float an = acosf(clip_ac(gn0 * n0 + gn1 * n1 + gn2 * n2));

    // ring-difference (roll by 1 over sorted positions)
    int prev = (lane + 31) & 31;
    float pl0 = __shfl(l0, prev, 32), pl1 = __shfl(l1, prev, 32), pl2 = __shfl(l2, prev, 32);
    float pg0 = __shfl(gn0, prev, 32), pg1 = __shfl(gn1, prev, 32), pg2 = __shfl(gn2, prev, 32);

    float i0 = l0 - pl0, i1 = l1 - pl1, i2 = l2 - pl2;
    float ilen = sqrtf(i0 * i0 + i1 * i1 + i2 * i2);
    float iinv = (ilen > 0.0f) ? (1.0f / ilen) : 0.0f;
    float iu0 = i0 * iinv, iu1 = i1 * iinv, iu2 = i2 * iinv;

    float ia0 = iu0 * gn0 + iu1 * gn1 + iu2 * gn2;
    float ia1 = iu0 * pg0 + iu1 * pg1 + iu2 * pg2;
    float ia2 = acosf(clip_ac(gn0 * pg0 + gn1 * pg1 + gn2 * pg2));

    _Float16* o = X0 + ((size_t)g * KDIM + lane) * 8;
    o[0] = (_Float16)llen; o[1] = (_Float16)pia; o[2] = (_Float16)a0; o[3] = (_Float16)a1;
    o[4] = (_Float16)an;   o[5] = (_Float16)ia0; o[6] = (_Float16)ia1; o[7] = (_Float16)ia2;
}

// ---------------------------------------------------------------------------
// Kernel 2: fused 4-layer MLP + max over K, one workgroup (8 waves) per group.
// WMMA f16 in / f32 accumulate. Activations live in LDS, [col][chan] f16.
// ---------------------------------------------------------------------------
__device__ __forceinline__ v16h load_a16(const _Float16* p) {
    // A-matrix 16x32 f16 per-lane packing: halves 0..7 = K(+0..7), 8..15 = K(+16..23)
    v8h lo = *(const v8h*)p;
    v8h hi = *(const v8h*)(p + 16);
    v16h v;
    #pragma unroll
    for (int i = 0; i < 8; i++) { v[i] = lo[i]; v[8 + i] = hi[i]; }
    return v;
}

template <int CIN, int COUT>
__device__ __forceinline__ void mlp_layer(
    const _Float16* __restrict__ W, const float* __restrict__ bias,
    const _Float16* __restrict__ Xin, _Float16* __restrict__ Xout,
    int wave, int lane)
{
    const int hs = lane >> 4, l15 = lane & 15;
    const int ntiles = (COUT / 16) * 2;           // Mtiles x 2 Ntiles (32 cols)
    for (int t = wave; t < ntiles; t += 8) {
        const int tm = t >> 1, tn = t & 1;
        v8f c = {};
        #pragma unroll
        for (int kk = 0; kk < CIN; kk += 32) {
            v16h a = load_a16(W + (tm * 16 + l15) * CIN + kk + hs * 8);
            // B-matrix 32x16: same packing as A: halves 0..7 = K(+0..7), 8..15 = K(+16..23), +8 for lanes 16-31
            v16h b = load_a16(Xin + (tn * 16 + l15) * CIN + kk + hs * 8);
            c = __builtin_amdgcn_wmma_f32_16x16x32_f16(false, a, false, b,
                                                       (short)0, c, false, false);
        }
        const int col = tn * 16 + l15;
        const int mb  = tm * 16 + hs * 8;
        _Float16* o = Xout + col * COUT + mb;
        #pragma unroll
        for (int r = 0; r < 8; r++) {
            float y = c[r] + bias[mb + r];
            o[r] = (_Float16)fmaxf(y, 0.0f);
        }
    }
}

__global__ __launch_bounds__(256) void mlp_kernel(
    const _Float16* __restrict__ X0,   // [G][32][8]
    const _Float16* __restrict__ W1,   // [64][32] (K-padded)
    const _Float16* __restrict__ W2,   // [128][64]
    const _Float16* __restrict__ W3,   // [128][128]
    const _Float16* __restrict__ W4,   // [256][128]
    const float*    __restrict__ BIAS, // 64 | 128 | 128 | 256
    float*          __restrict__ out)  // [G][256]
{
    __shared__ __align__(32) _Float16 A1[32 * 64];
    __shared__ __align__(32) _Float16 A2[32 * 128];
    __shared__ __align__(32) _Float16 A3[32 * 128];
    __shared__ float Ymax[256];

    const int tid = threadIdx.x, lane = tid & 31, wave = tid >> 5;
    const int hs = lane >> 4, l15 = lane & 15;
    const int g = blockIdx.x;

    Ymax[tid] = 0.0f;   // ReLU outputs are >= 0, so 0 is the identity for the max

    const float* bias1 = BIAS;
    const float* bias2 = BIAS + 64;
    const float* bias3 = BIAS + 192;
    const float* bias4 = BIAS + 320;

    // ---- Layer 1: 64 x 8(pad 32) x 32 cols, 8 tiles, one per wave -------
    {
        const int tm = wave >> 1, tn = wave & 1;
        v16h a = load_a16(W1 + (tm * 16 + l15) * 32 + hs * 8);
        v16h b;
        #pragma unroll
        for (int i = 0; i < 16; i++) b[i] = (_Float16)0;
        if (hs == 0) {  // K 0..7 are the 8 real input channels; rest zero-padded
            v8h x = *(const v8h*)(X0 + ((size_t)g * 32 + tn * 16 + l15) * 8);
            #pragma unroll
            for (int i = 0; i < 8; i++) b[i] = x[i];
        }
        v8f c = {};
        c = __builtin_amdgcn_wmma_f32_16x16x32_f16(false, a, false, b,
                                                   (short)0, c, false, false);
        const int col = tn * 16 + l15;
        const int mb  = tm * 16 + hs * 8;
        _Float16* o = A1 + col * 64 + mb;
        #pragma unroll
        for (int r = 0; r < 8; r++) {
            float y = c[r] + bias1[mb + r];
            o[r] = (_Float16)fmaxf(y, 0.0f);
        }
    }
    __syncthreads();

    mlp_layer<64, 128>(W2, bias2, A1, A2, wave, lane);   // Layer 2
    __syncthreads();
    mlp_layer<128, 128>(W3, bias3, A2, A3, wave, lane);  // Layer 3
    __syncthreads();

    // ---- Layer 4: 256 x 128 x 32, 32 tiles, fused max over the 32 cols --
    for (int t = wave; t < 32; t += 8) {
        const int tm = t >> 1, tn = t & 1;
        v8f c = {};
        #pragma unroll
        for (int kk = 0; kk < 128; kk += 32) {
            v16h a = load_a16(W4 + (tm * 16 + l15) * 128 + kk + hs * 8);
            v16h b = load_a16(A3 + (tn * 16 + l15) * 128 + kk + hs * 8);
            c = __builtin_amdgcn_wmma_f32_16x16x32_f16(false, a, false, b,
                                                       (short)0, c, false, false);
        }
        const int mb = tm * 16 + hs * 8;
        #pragma unroll
        for (int r = 0; r < 8; r++) {
            float y = fmaxf(c[r] + bias4[mb + r], 0.0f);
            // non-negative floats order identically to their int bit patterns
            atomicMax((int*)&Ymax[mb + r], __float_as_int(y));
        }
    }
    __syncthreads();

    out[(size_t)g * 256 + tid] = Ymax[tid];
}

// ---------------------------------------------------------------------------
// Launch
// ---------------------------------------------------------------------------
extern "C" void kernel_launch(void* const* d_in, const int* in_sizes, int n_in,
                              void* d_out, int out_size, void* d_ws, size_t ws_size,
                              hipStream_t stream)
{
    (void)in_sizes; (void)n_in; (void)out_size; (void)ws_size;

    const float* xyz     = (const float*)d_in[0];
    const float* centers = (const float*)d_in[1];
    // d_in[2] = norm : unused by the reference computation
    const float* cnorms  = (const float*)d_in[3];
    const int*   idx     = (const int*)  d_in[4];
    const float* pts     = (const float*)d_in[5];
    const float* normori = (const float*)d_in[6];
    const float* w1 = (const float*)d_in[7],  *b1 = (const float*)d_in[8],
               * g1 = (const float*)d_in[9],  *bt1 = (const float*)d_in[10];
    const float* w2 = (const float*)d_in[11], *b2 = (const float*)d_in[12],
               * g2 = (const float*)d_in[13], *bt2 = (const float*)d_in[14];
    const float* w3 = (const float*)d_in[15], *b3 = (const float*)d_in[16],
               * g3 = (const float*)d_in[17], *bt3 = (const float*)d_in[18];
    const float* w4 = (const float*)d_in[19], *b4 = (const float*)d_in[20],
               * g4 = (const float*)d_in[21], *bt4 = (const float*)d_in[22];

    // workspace layout (f16 weights, f32 biases, f16 staged features) ~4.3 MB
    _Float16* W1 = (_Float16*)d_ws;          //  64*32  = 2048 halves
    _Float16* W2 = W1 + 64 * 32;             // 128*64  = 8192
    _Float16* W3 = W2 + 128 * 64;            // 128*128 = 16384
    _Float16* W4 = W3 + 128 * 128;           // 256*128 = 32768
    float*    BIAS = (float*)(W4 + 256 * 128);          // 576 floats
    _Float16* X0 = (_Float16*)(BIAS + 576);             // 8192*32*8 halves

    prep_kernel<<<64, 256, 0, stream>>>(w1, b1, g1, bt1, w2, b2, g2, bt2,
                                        w3, b3, g3, bt3, w4, b4, g4, bt4,
                                        W1, W2, W3, W4, BIAS);

    geom_kernel<<<NGRP, 32, 0, stream>>>(xyz, centers, cnorms, idx, pts, normori, X0);

    mlp_kernel<<<NGRP, 256, 0, stream>>>(X0, W1, W2, W3, W4, BIAS, (float*)d_out);
}
